// GraphMamba_31997506356068
// MI455X (gfx1250) — matrix operand order repaired and checked by reading, not compiled, not run on hardware
//
#include <hip/hip_runtime.h>
#include <hip/hip_bf16.h>

typedef __attribute__((ext_vector_type(16))) _Float16 v16h;
typedef __attribute__((ext_vector_type(8)))  float    v8f;
typedef __attribute__((ext_vector_type(4)))  unsigned int u32x4;
typedef __attribute__((ext_vector_type(8)))  int      i32x8;
typedef __attribute__((ext_vector_type(4)))  int      i32x4;

#define D_   256
#define DI_  512
#define DS_  16
#define NCH  128
#define MAX_KSTEPS 24   // K<=768

// ---------- helpers ----------
__device__ __forceinline__ float silu_f(float x) { return x / (1.0f + __expf(-x)); }
__device__ __forceinline__ unsigned fenc(float f) {
  unsigned u = __float_as_uint(f);
  return (u & 0x80000000u) ? ~u : (u | 0x80000000u);
}
__device__ __forceinline__ float fdec(unsigned u) {
  u = (u & 0x80000000u) ? (u & 0x7FFFFFFFu) : ~u;
  return __uint_as_float(u);
}

__device__ __forceinline__ v16h load_a_frag(const float* p0, const float* p1) {
  float4 q0 = *(const float4*)(p0);
  float4 q1 = *(const float4*)(p0 + 4);
  float4 q2 = *(const float4*)(p1);
  float4 q3 = *(const float4*)(p1 + 4);
  v16h a;
  a[0]=(_Float16)q0.x; a[1]=(_Float16)q0.y; a[2]=(_Float16)q0.z; a[3]=(_Float16)q0.w;
  a[4]=(_Float16)q1.x; a[5]=(_Float16)q1.y; a[6]=(_Float16)q1.z; a[7]=(_Float16)q1.w;
  a[8]=(_Float16)q2.x; a[9]=(_Float16)q2.y; a[10]=(_Float16)q2.z; a[11]=(_Float16)q2.w;
  a[12]=(_Float16)q3.x; a[13]=(_Float16)q3.y; a[14]=(_Float16)q3.z; a[15]=(_Float16)q3.w;
  return a;
}

// ---------- weight pre-swizzle into B-operand lane layout ----------
// B 32x16 f16: lane half h holds K=8h+e (e<8) and K=16+8h+(e-8) (e>=8) at col N=lane%16.
__global__ void preswz_kernel(const float* __restrict__ w, _Float16* __restrict__ o,
                              int K, int Nc) {
  int t = blockIdx.x * blockDim.x + threadIdx.x;
  int ksteps = K >> 5;
  int total = (Nc >> 4) * ksteps * 32;
  if (t >= total) return;
  int lane  = t & 31;
  int ks    = (t >> 5) % ksteps;
  int ntile = (t >> 5) / ksteps;
  int hi    = lane >> 4;
  int n     = ntile * 16 + (lane & 15);
  _Float16* dst = o + (size_t)t * 16;
#pragma unroll
  for (int e = 0; e < 16; ++e) {
    int k = (ks << 5) + (e & 7) + hi * 8 + (e >> 3) * 16;
    dst[e] = (_Float16)w[(size_t)k * Nc + n];
  }
}

// ---------- generic WMMA GEMM: C[M x Nc] = A[M x K] * Wswz (+bias), opt. row scatter ----------
// The per-n_tile B panel (ksteps*1024 bytes) is DMA'd to LDS once per block via the
// Tensor Data Mover, then all 8 waves feed WMMA from LDS.
// HOP mode: A is the virtual hop-concat of x[3][hop_n][256]; k = h*256+d.
template<bool HOP>
__global__ __launch_bounds__(256)
void wmma_gemm_kernel(const float* __restrict__ A, const _Float16* __restrict__ Bswz,
                      float* __restrict__ C, const float* __restrict__ bias,
                      const int* __restrict__ row_map,
                      int M, int K, int Nc, int lda, int ldc, int hop_n) {
  __shared__ _Float16 Bsh[MAX_KSTEPS * 512];   // 24 KB max panel
  const int lane = threadIdx.x & 31;
  const int wave = threadIdx.x >> 5;
  const int m0 = blockIdx.x * 128 + wave * 16;
  const int n_tile = blockIdx.y;
  const int n0 = n_tile * 16;
  const int r  = lane & 15;
  const int hi = lane >> 4;
  const int ksteps = K >> 5;

  // --- TDM: async-copy the B panel for this n_tile into LDS (wave 0 only) ---
  if (wave == 0) {
    const unsigned nb8 = (unsigned)(ksteps * 128);  // panel size in 8-byte elements
    unsigned long long ga =
        (unsigned long long)(uintptr_t)(Bswz + (size_t)n_tile * ksteps * 512);
    unsigned lds = (unsigned)(uintptr_t)&Bsh[0];
    u32x4 g0;
    g0[0] = 1u;                                          // count=1, user mode
    g0[1] = lds;                                         // lds_addr
    g0[2] = (unsigned)(ga & 0xFFFFFFFFu);                // global_addr[31:0]
    g0[3] = (unsigned)((ga >> 32) & 0x1FFFFFFu) | (2u << 30);  // addr hi | type=2
    i32x8 g1;
    g1[0] = (int)(3u << 16);                             // data_size = 8B
    g1[1] = (int)((nb8 & 0xFFFFu) << 16);                // tensor_dim0[15:0]
    g1[2] = (int)(((nb8 >> 16) & 0xFFFFu) | (1u << 16)); // td0[31:16] | tensor_dim1=1
    g1[3] = (int)((nb8 & 0xFFFFu) << 16);                // tile_dim0 = nb8
    g1[4] = 0;                                           // tile_dim1/2 unused
    g1[5] = (int)nb8;                                    // tensor_dim0_stride lo
    g1[6] = 0;
    g1[7] = 0;
    i32x4 gz; gz[0]=0; gz[1]=0; gz[2]=0; gz[3]=0;
#if __clang_major__ >= 23
    i32x8 g4z; g4z[0]=0; g4z[1]=0; g4z[2]=0; g4z[3]=0; g4z[4]=0; g4z[5]=0; g4z[6]=0; g4z[7]=0;
    __builtin_amdgcn_tensor_load_to_lds(g0, g1, gz, gz, g4z, 0);
#else
    __builtin_amdgcn_tensor_load_to_lds(g0, g1, gz, gz, 0);
#endif
    __builtin_amdgcn_s_wait_tensorcnt(0);
  }
  __syncthreads();

  v8f acc = {0.f,0.f,0.f,0.f,0.f,0.f,0.f,0.f};
  int m = m0 + r; if (m >= M) m = M - 1;   // clamp (tail safety)

  const _Float16* bls = Bsh + ((size_t)lane << 4);

  if (HOP) {
#pragma unroll 2
    for (int ks = 0; ks < ksteps; ++ks) {
      int kb = (ks << 5) + hi * 8;
      int k1 = kb + 16;
      const float* p0 = A + ((size_t)(kb >> 8) * hop_n + m) * 256 + (kb & 255);
      const float* p1 = A + ((size_t)(k1 >> 8) * hop_n + m) * 256 + (k1 & 255);
      __builtin_prefetch(p0 + 32, 0, 1);
      v16h a = load_a_frag(p0, p1);
      v16h b = *(const v16h*)(bls + ((size_t)ks << 9));
      acc = __builtin_amdgcn_wmma_f32_16x16x32_f16(false, a, false, b,
                                                   (short)0, acc, false, false);
    }
  } else {
    const float* p0 = A + (size_t)m * lda + hi * 8;
    const float* p1 = p0 + 16;
#pragma unroll 2
    for (int ks = 0; ks < ksteps; ++ks) {
      __builtin_prefetch(p0 + 32, 0, 1);
      v16h a = load_a_frag(p0, p1);
      p0 += 32; p1 += 32;
      v16h b = *(const v16h*)(bls + ((size_t)ks << 9));
      acc = __builtin_amdgcn_wmma_f32_16x16x32_f16(false, a, false, b,
                                                   (short)0, acc, false, false);
    }
  }

  float bv = bias ? bias[n0 + r] : 0.0f;
  if (m0 + 16 <= M) {           // full tile (always true when M % 128 == 0)
#pragma unroll
    for (int i = 0; i < 8; ++i) {
      int mi = m0 + hi * 8 + i;
      int row = row_map ? row_map[mi] : mi;
      C[(size_t)row * ldc + n0 + r] = acc[i] + bv;
    }
  } else {
#pragma unroll
    for (int i = 0; i < 8; ++i) {
      int mi = m0 + hi * 8 + i;
      if (mi < M) {
        int row = row_map ? row_map[mi] : mi;
        C[(size_t)row * ldc + n0 + r] = acc[i] + bv;
      }
    }
  }
}

// ---------- LayerNorm(256) + SiLU, in place ----------
__global__ __launch_bounds__(256)
void ln_silu_kernel(float* __restrict__ X, const float* __restrict__ g,
                    const float* __restrict__ b) {
  __shared__ float red[256];
  int row = blockIdx.x, tid = threadIdx.x;
  float v = X[(size_t)row * 256 + tid];
  red[tid] = v; __syncthreads();
  for (int s = 128; s > 0; s >>= 1) { if (tid < s) red[tid] += red[tid + s]; __syncthreads(); }
  float mean = red[0] * (1.0f / 256.0f); __syncthreads();
  float d = v - mean;
  red[tid] = d * d; __syncthreads();
  for (int s = 128; s > 0; s >>= 1) { if (tid < s) red[tid] += red[tid + s]; __syncthreads(); }
  float var = red[0] * (1.0f / 256.0f);
  float y = d * rsqrtf(var + 1e-5f) * g[tid] + b[tid];
  X[(size_t)row * 256 + tid] = silu_f(y);
}

// ---------- init: cat[:, :256]=gat_b, emax/denom, sort keys/idx ----------
__global__ void init_kernel(float* __restrict__ cat, const float* __restrict__ gat_b,
                            unsigned* __restrict__ emax, float* __restrict__ denom,
                            float* __restrict__ keys, int* __restrict__ idxs,
                            const float* __restrict__ noise, int N) {
  int n = blockIdx.x * blockDim.x + threadIdx.x;
  if (n >= N) return;
  emax[n] = 0u; denom[n] = 0.0f;
  keys[n] = 0.1f * noise[n]; idxs[n] = n;
  float* c = cat + (size_t)n * 512;
  for (int d = 0; d < 256; ++d) c[d] = gat_b[d];
}

// ---------- GAT: per-node src/dst attention scores ----------
__global__ void gat_score_kernel(const float* __restrict__ h, const float* __restrict__ asrc,
                                 const float* __restrict__ adst, float* __restrict__ s,
                                 float* __restrict__ t, int N) {
  int n = blockIdx.x * blockDim.x + threadIdx.x;
  if (n >= N) return;
  const float* hp = h + (size_t)n * 256;
  float ss = 0.f, tt = 0.f;
  for (int d = 0; d < 256; d += 4) {
    float4 v = *(const float4*)(hp + d);
    ss += v.x*asrc[d] + v.y*asrc[d+1] + v.z*asrc[d+2] + v.w*asrc[d+3];
    tt += v.x*adst[d] + v.y*adst[d+1] + v.z*adst[d+2] + v.w*adst[d+3];
  }
  s[n] = ss; t[n] = tt;
}

__global__ void edge_score_kernel(const int* __restrict__ ei, const float* __restrict__ s,
                                  const float* __restrict__ t, float* __restrict__ ev,
                                  unsigned* __restrict__ emax, int E, int N) {
  int e = blockIdx.x * blockDim.x + threadIdx.x;
  if (e >= E + N) return;
  int src = (e < E) ? ei[e]     : (e - E);
  int dst = (e < E) ? ei[E + e] : (e - E);
  float v = s[src] + t[dst];
  v = (v > 0.f) ? v : 0.2f * v;          // leaky_relu 0.2
  ev[e] = v;
  atomicMax(&emax[dst], fenc(v));
}

__global__ void edge_exp_kernel(const int* __restrict__ ei, float* __restrict__ ev,
                                const unsigned* __restrict__ emax, float* __restrict__ denom,
                                int E, int N) {
  int e = blockIdx.x * blockDim.x + threadIdx.x;
  if (e >= E + N) return;
  int dst = (e < E) ? ei[E + e] : (e - E);
  float x = __expf(ev[e] - fdec(emax[dst]));
  ev[e] = x;
  atomicAdd(&denom[dst], x);
}

__global__ __launch_bounds__(256)
void edge_scatter_kernel(const int* __restrict__ ei, const float* __restrict__ ev,
                         const float* __restrict__ denom, const float* __restrict__ h,
                         float* __restrict__ cat, int E) {
  int e = blockIdx.x;
  int d = threadIdx.x;
  int N_off = e - E;
  int src = (e < E) ? ei[e]     : N_off;
  int dst = (e < E) ? ei[E + e] : N_off;
  float alpha = ev[e] / denom[dst];
  atomicAdd(&cat[(size_t)dst * 512 + d], alpha * h[(size_t)src * 256 + d]);
}

__global__ void deg_kernel(const int* __restrict__ ei, float* __restrict__ keys, int E) {
  int e = blockIdx.x * blockDim.x + threadIdx.x;
  if (e < E) atomicAdd(&keys[ei[e]], 1.0f);
}

// ---------- bitonic sort pass (keys asc, idx tiebreak) ----------
__global__ void bitonic_kernel(float* __restrict__ keys, int* __restrict__ idxs,
                               int j, int k, int N) {
  int i = blockIdx.x * blockDim.x + threadIdx.x;
  if (i >= N) return;
  int l = i ^ j;
  if (l <= i) return;
  float ki = keys[i], kl = keys[l];
  int ii = idxs[i], il = idxs[l];
  bool asc = ((i & k) == 0);
  bool gt = (ki > kl) || (ki == kl && ii > il);
  if (asc ? gt : !gt) { keys[i] = kl; keys[l] = ki; idxs[i] = il; idxs[l] = ii; }
}

__global__ __launch_bounds__(256)
void gather_kernel(const float* __restrict__ xf, const int* __restrict__ idxs,
                   float* __restrict__ xs) {
  int r = blockIdx.x, d = threadIdx.x;
  xs[(size_t)r * 256 + d] = xf[(size_t)idxs[r] * 256 + d];
}

// ---------- depthwise causal conv (k=4) + SiLU ----------
__global__ void conv_silu_kernel(const float* __restrict__ xz, const float* __restrict__ cw,
                                 float* __restrict__ xm, int N) {
  int id = blockIdx.x * blockDim.x + threadIdx.x;
  if (id >= N * DI_) return;
  int t = id >> 9, d = id & 511;
  float acc = 0.f;
#pragma unroll
  for (int j = 0; j < 4; ++j) {
    int tt = t - 3 + j;
    if (tt >= 0) acc += cw[d * 4 + j] * xz[(size_t)tt * 1024 + d];
  }
  xm[(size_t)t * DI_ + d] = silu_f(acc);
}

// ---------- x_dbl = xm @ x_proj_w (512x17) ----------
__global__ __launch_bounds__(32)
void xdbl_kernel(const float* __restrict__ xm, const float* __restrict__ xpw,
                 float* __restrict__ xdbl) {
  __shared__ float row[DI_];
  int t = blockIdx.x;
  for (int d = threadIdx.x; d < DI_; d += 32) row[d] = xm[(size_t)t * DI_ + d];
  __syncthreads();
  int c = threadIdx.x;
  if (c < 17) {
    float acc = 0.f;
    for (int d = 0; d < DI_; ++d) acc += row[d] * xpw[d * 17 + c];
    xdbl[t * 17 + c] = acc;
  }
}

__device__ __forceinline__ float softplus_f(float x) {
  return (x > 20.f) ? x : log1pf(__expf(x));
}

// ---------- chunked selective scan: pass1 (local), pass2 (compose), pass3 (correct+gate) ----------
__global__ __launch_bounds__(512)
void scan_pass1_kernel(const float* __restrict__ xdbl, const float* __restrict__ dtw,
                       const float* __restrict__ dtb, const float* __restrict__ alog,
                       float* __restrict__ yb, float* __restrict__ csum,
                       float* __restrict__ chh, int N) {
  int c = blockIdx.x, d = threadIdx.x;
  int L = N / NCH, t0 = c * L;
  float An[DS_], h[DS_];
#pragma unroll
  for (int n = 0; n < DS_; ++n) { An[n] = -__expf(alog[d * DS_ + n]); h[n] = 0.f; }
  float wdt = dtw[d], bdt = dtb[d], s = 0.f;
  for (int t = t0; t < t0 + L; ++t) {
    const float* xr = xdbl + t * 17;
    float dt = softplus_f(xr[0] * wdt + bdt);
    s += dt;
    float y = 0.f;
#pragma unroll
    for (int n = 0; n < DS_; ++n) {
      float Bn = xr[1 + n];
      h[n] = __expf(An[n] * dt) * h[n] + dt * Bn;
      y += h[n] * Bn;
    }
    yb[(size_t)t * 1024 + d] = y;                 // local y (init state 0)
  }
  csum[c * DI_ + d] = s;
#pragma unroll
  for (int n = 0; n < DS_; ++n) chh[((size_t)c * DI_ + d) * DS_ + n] = h[n];
}

__global__ __launch_bounds__(512)
void scan_pass2_kernel(const float* __restrict__ csum, const float* __restrict__ chh,
                       const float* __restrict__ alog, float* __restrict__ hinit) {
  int d = threadIdx.x;
  float An[DS_], h[DS_];
#pragma unroll
  for (int n = 0; n < DS_; ++n) { An[n] = -__expf(alog[d * DS_ + n]); h[n] = 0.f; }
  for (int c = 0; c < NCH; ++c) {
#pragma unroll
    for (int n = 0; n < DS_; ++n) hinit[((size_t)c * DI_ + d) * DS_ + n] = h[n];
    float s = csum[c * DI_ + d];
#pragma unroll
    for (int n = 0; n < DS_; ++n)
      h[n] = __expf(An[n] * s) * h[n] + chh[((size_t)c * DI_ + d) * DS_ + n];
  }
}

__global__ __launch_bounds__(512)
void scan_pass3_kernel(const float* __restrict__ xdbl, const float* __restrict__ dtw,
                       const float* __restrict__ dtb, const float* __restrict__ alog,
                       const float* __restrict__ hinit, const float* __restrict__ xm,
                       const float* __restrict__ Dp, float* __restrict__ xz, int N) {
  int c = blockIdx.x, d = threadIdx.x;
  int L = N / NCH, t0 = c * L;
  float An[DS_], h0[DS_];
#pragma unroll
  for (int n = 0; n < DS_; ++n) {
    An[n] = -__expf(alog[d * DS_ + n]);
    h0[n] = hinit[((size_t)c * DI_ + d) * DS_ + n];
  }
  float wdt = dtw[d], bdt = dtb[d], Dd = Dp[d], s = 0.f;
  for (int t = t0; t < t0 + L; ++t) {
    const float* xr = xdbl + t * 17;
    float dt = softplus_f(xr[0] * wdt + bdt);
    s += dt;
    float corr = 0.f;
#pragma unroll
    for (int n = 0; n < DS_; ++n) corr += __expf(An[n] * s) * h0[n] * xr[1 + n];
    float y = xz[(size_t)t * 1024 + d] + corr;          // local + init-state correction
    y += xm[(size_t)t * DI_ + d] * Dd;
    float z = xz[(size_t)t * 1024 + 512 + d];
    xz[(size_t)t * 1024 + d] = y * silu_f(z);           // gated output, in place
  }
}

// ---------- launch ----------
extern "C" void kernel_launch(void* const* d_in, const int* in_sizes, int n_in,
                              void* d_out, int out_size, void* d_ws, size_t ws_size,
                              hipStream_t stream) {
  const float* x        = (const float*)d_in[0];
  const int*   ei       = (const int*)  d_in[1];
  const float* noise    = (const float*)d_in[2];
  const float* hop_w    = (const float*)d_in[3];
  const float* hop_b    = (const float*)d_in[4];
  const float* hop_ln_g = (const float*)d_in[5];
  const float* hop_ln_b = (const float*)d_in[6];
  const float* gat_w    = (const float*)d_in[7];
  const float* att_src  = (const float*)d_in[8];
  const float* att_dst  = (const float*)d_in[9];
  const float* gat_b    = (const float*)d_in[10];
  const float* inp_w    = (const float*)d_in[11];
  const float* conv_w   = (const float*)d_in[12];
  const float* xproj_w  = (const float*)d_in[13];
  const float* dt_w     = (const float*)d_in[14];
  const float* dt_b     = (const float*)d_in[15];
  const float* A_log    = (const float*)d_in[16];
  const float* D_param  = (const float*)d_in[17];
  const float* out_w    = (const float*)d_in[18];
  const float* out_b    = (const float*)d_in[19];
  const float* mlp_w1   = (const float*)d_in[20];
  const float* mlp_b1   = (const float*)d_in[21];
  const float* mlp_ln_g = (const float*)d_in[22];
  const float* mlp_ln_b = (const float*)d_in[23];
  const float* mlp_w2   = (const float*)d_in[24];
  const float* mlp_b2   = (const float*)d_in[25];
  float* out = (float*)d_out;

  const int N = in_sizes[2];          // 16384
  const int E = in_sizes[1] / 2;      // 262144
  const int EN = E + N;

  // workspace carve-out
  char* base = (char*)d_ws; size_t off = 0;
  auto alloc = [&](size_t bytes) -> void* {
    void* p = base + off; off = (off + bytes + 255) & ~(size_t)255; return p;
  };
  _Float16* swz_hop = (_Float16*)alloc((size_t)768 * 256 * 2);
  _Float16* swz_gat = (_Float16*)alloc((size_t)256 * 256 * 2);
  _Float16* swz_inp = (_Float16*)alloc((size_t)256 * 1024 * 2);
  _Float16* swz_out = (_Float16*)alloc((size_t)512 * 256 * 2);
  _Float16* swz_m1  = (_Float16*)alloc((size_t)512 * 256 * 2);
  _Float16* swz_m2  = (_Float16*)alloc((size_t)256 * 256 * 2);
  float*    xf      = (float*)alloc((size_t)N * 256 * 4);
  float*    hbuf    = (float*)alloc((size_t)N * 256 * 4);
  float*    sbuf    = (float*)alloc((size_t)N * 4);
  float*    tbuf    = (float*)alloc((size_t)N * 4);
  float*    evbuf   = (float*)alloc((size_t)EN * 4);
  unsigned* emax    = (unsigned*)alloc((size_t)N * 4);
  float*    denom   = (float*)alloc((size_t)N * 4);
  float*    keys    = (float*)alloc((size_t)N * 4);
  int*      idxs    = (int*)alloc((size_t)N * 4);
  float*    cat     = (float*)alloc((size_t)N * 512 * 4);
  float*    xs      = (float*)alloc((size_t)N * 256 * 4);  // reused as MLP hidden t1
  float*    xz      = (float*)alloc((size_t)N * 1024 * 4); // first half reused as y
  float*    xm      = (float*)alloc((size_t)N * 512 * 4);
  float*    xdbl    = (float*)alloc((size_t)N * 17 * 4);
  float*    csum    = (float*)alloc((size_t)NCH * 512 * 4);
  float*    chh     = (float*)alloc((size_t)NCH * 512 * 16 * 4);
  float*    hinit   = (float*)alloc((size_t)NCH * 512 * 16 * 4);
  (void)ws_size;

  auto swz_launch = [&](const float* w, _Float16* o, int K, int Nc) {
    int total = (Nc / 16) * (K / 32) * 32;
    preswz_kernel<<<(total + 255) / 256, 256, 0, stream>>>(w, o, K, Nc);
  };
  swz_launch(hop_w, swz_hop, 768, 256);
  swz_launch(gat_w, swz_gat, 256, 256);
  swz_launch(inp_w, swz_inp, 256, 1024);
  swz_launch(out_w, swz_out, 512, 256);
  swz_launch(mlp_w1, swz_m1, 512, 256);
  swz_launch(mlp_w2, swz_m2, 256, 256);

  init_kernel<<<(N + 255) / 256, 256, 0, stream>>>(cat, gat_b, emax, denom, keys, idxs, noise, N);

  const int mblk = (N + 127) / 128;
  auto gemm = [&](const float* A, const _Float16* B, float* C, const float* bias,
                  const int* rmap, int K, int Nc, int lda, int ldc) {
    dim3 grid(mblk, Nc / 16);
    wmma_gemm_kernel<false><<<grid, 256, 0, stream>>>(A, B, C, bias, rmap,
                                                      N, K, Nc, lda, ldc, 0);
  };

  // (1) hop projection + LN + SiLU -> xf
  {
    dim3 grid(mblk, 256 / 16);
    wmma_gemm_kernel<true><<<grid, 256, 0, stream>>>(x, swz_hop, xf, hop_b, nullptr,
                                                     N, 768, 256, 0, 256, N);
  }
  ln_silu_kernel<<<N, 256, 0, stream>>>(xf, hop_ln_g, hop_ln_b);

  // (2) GAT
  gemm(xf, swz_gat, hbuf, nullptr, nullptr, 256, 256, 256, 256);
  gat_score_kernel<<<(N + 255) / 256, 256, 0, stream>>>(hbuf, att_src, att_dst, sbuf, tbuf, N);
  edge_score_kernel<<<(EN + 255) / 256, 256, 0, stream>>>(ei, sbuf, tbuf, evbuf, emax, E, N);
  edge_exp_kernel<<<(EN + 255) / 256, 256, 0, stream>>>(ei, evbuf, emax, denom, E, N);
  edge_scatter_kernel<<<EN, 256, 0, stream>>>(ei, evbuf, denom, hbuf, cat, E);

  // (3) degree + argsort (bitonic, N is a power of two)
  deg_kernel<<<(E + 255) / 256, 256, 0, stream>>>(ei, keys, E);
  for (int k = 2; k <= N; k <<= 1)
    for (int j = k >> 1; j > 0; j >>= 1)
      bitonic_kernel<<<(N + 255) / 256, 256, 0, stream>>>(keys, idxs, j, k, N);

  // (4) Mamba
  gather_kernel<<<N, 256, 0, stream>>>(xf, idxs, xs);
  gemm(xs, swz_inp, xz, nullptr, nullptr, 256, 1024, 256, 1024);
  conv_silu_kernel<<<(N * 512 + 255) / 256, 256, 0, stream>>>(xz, conv_w, xm, N);
  xdbl_kernel<<<N, 32, 0, stream>>>(xm, xproj_w, xdbl);
  scan_pass1_kernel<<<NCH, 512, 0, stream>>>(xdbl, dt_w, dt_b, A_log, xz, csum, chh, N);
  scan_pass2_kernel<<<1, 512, 0, stream>>>(csum, chh, A_log, hinit);
  scan_pass3_kernel<<<NCH, 512, 0, stream>>>(xdbl, dt_w, dt_b, A_log, hinit, xm, D_param, xz, N);
  // out-proj with fused scatter back to original node order: cat[:, 256:512]
  gemm(xz, swz_out, cat + 256, out_b, idxs, 512, 256, 1024, 512);

  // (5) MLP head
  gemm(cat, swz_m1, xs, mlp_b1, nullptr, 512, 256, 512, 256);
  ln_silu_kernel<<<N, 256, 0, stream>>>(xs, mlp_ln_g, mlp_ln_b);
  gemm(xs, swz_m2, out, mlp_b2, nullptr, 256, 256, 256, 256);
  (void)out_size; (void)n_in;
}
